// Transformer_20753281974613
// MI455X (gfx1250) — matrix operand order repaired and checked by reading, not compiled, not run on hardware
//
#include <hip/hip_runtime.h>
#include <hip/hip_bf16.h>
#include <cstddef>

// ---------------------------------------------------------------------------
// MI455X (gfx1250) implementation.
// GEMMs (~446 GFLOP, dominant) and both attention einsums run on
// v_wmma_f32_16x16x32_bf16 with fp32 accumulation; fp32->bf16 conversion is
// done while staging tiles into LDS. The GEMM is double-buffered (ping-pong
// LDS, register-staged global loads) so global latency hides under WMMA.
// Norms / rope / softmax bookkeeping / LoRA rank-16 are fp32 VALU (<5%).
// ---------------------------------------------------------------------------

typedef __attribute__((ext_vector_type(16))) __bf16 v16bf;
typedef __attribute__((ext_vector_type(8)))  __bf16 v8bf;
typedef __attribute__((ext_vector_type(8)))  float  v8f;

__device__ __forceinline__ unsigned short f2bf(float f) {
    unsigned u = __builtin_bit_cast(unsigned, f);
    unsigned r = u + 0x7FFFu + ((u >> 16) & 1u);   // round-to-nearest-even
    return (unsigned short)(r >> 16);
}
__device__ __forceinline__ unsigned pack_bf2(float a, float b) {
    return (unsigned)f2bf(a) | ((unsigned)f2bf(b) << 16);
}
__device__ __forceinline__ __bf16 bfc(float f) {
    return __builtin_bit_cast(__bf16, f2bf(f));
}

// ---------------------------------------------------------------------------
// RMSNorm: one block per row (D=4096), 256 threads.
// ---------------------------------------------------------------------------
__global__ __launch_bounds__(256) void rmsnorm_kernel(
    const float* __restrict__ x, const float* __restrict__ w,
    float* __restrict__ out, int D)
{
    __shared__ float red[256];
    const int row = blockIdx.x;
    const int tid = threadIdx.x;
    const float* xr = x + (size_t)row * D;
    float s = 0.f;
    for (int i = tid; i < D; i += 256) { float v = xr[i]; s = fmaf(v, v, s); }
    red[tid] = s;
    __syncthreads();
    for (int off = 128; off > 0; off >>= 1) {
        if (tid < off) red[tid] += red[tid + off];
        __syncthreads();
    }
    const float rs = rsqrtf(red[0] / (float)D + 1e-5f);
    float* orow = out + (size_t)row * D;
    for (int i = tid; i < D; i += 256) orow[i] = xr[i] * rs * w[i];
}

// ---------------------------------------------------------------------------
// LoRA A-projection: t[m, r] = sum_k x[m,k] * A[r,k];  R = 16.
// ---------------------------------------------------------------------------
__global__ __launch_bounds__(256) void lora_a_kernel(
    const float* __restrict__ x, const float* __restrict__ A,
    float* __restrict__ t, int K)
{
    const int tid = threadIdx.x;
    const int m = blockIdx.x * 16 + (tid >> 4);
    const int r = tid & 15;
    const float4* xr = (const float4*)(x + (size_t)m * K);
    const float4* ar = (const float4*)(A + (size_t)r * K);
    float acc = 0.f;
    const int K4 = K >> 2;
    for (int i = 0; i < K4; ++i) {
        float4 xa = xr[i], aa = ar[i];
        acc = fmaf(xa.x, aa.x, acc);
        acc = fmaf(xa.y, aa.y, acc);
        acc = fmaf(xa.z, aa.z, acc);
        acc = fmaf(xa.w, aa.w, acc);
    }
    t[(size_t)m * 16 + r] = acc;
}

// ---------------------------------------------------------------------------
// Tiled bf16-WMMA GEMM, double-buffered, with fused LoRA epilogue:
//   out[M,N] = X[M,K] @ W[N,K]^T + 2.0 * (T[M,16] @ LB[N,16]^T)  (+ epilogue)
// MODE 0: store   MODE 1: += EX (residual)   MODE 2: *= silu(EX) (gate fuse)
// Block tile 128x64, BK=32, 8 waves, 2x2 WMMA subtiles/wave, ping-pong LDS.
// ---------------------------------------------------------------------------
#define BM 128
#define BN 64
#define BK 32

template<int MODE>
__global__ __launch_bounds__(256) void gemm_lora_kernel(
    const float* __restrict__ X, const float* __restrict__ W,
    const float* __restrict__ T, const float* __restrict__ LB,
    const float* __restrict__ EX, float* __restrict__ out,
    int M, int N, int K)
{
    __shared__ __align__(32) __bf16 lds_a[2][BM * BK];   // 2 x 8 KB
    __shared__ __align__(32) __bf16 lds_b[2][BN * BK];   // 2 x 4 KB

    const int tid  = threadIdx.x;
    const int wave = tid >> 5;
    const int lane = tid & 31;
    const int wm = wave & 3;       // 4 waves along M
    const int wn = wave >> 2;      // 2 waves along N
    const int ln  = lane & 15;
    const int lhi = lane >> 4;
    const int m0 = blockIdx.y * BM;
    const int n0 = blockIdx.x * BN;
    const int klo  = lhi ? 8 : 0;
    const int koff = lhi ? 16 : 0;

    // Per-thread staging coordinates (same for both A and B streams).
    const int sr = tid >> 3;        // 0..31  (row step 32 for both A and B)
    const int sc = (tid & 7) * 4;   // column within BK

    v8f acc[2][2];
    const v8f vzero = {0.f, 0.f, 0.f, 0.f, 0.f, 0.f, 0.f, 0.f};
    acc[0][0] = vzero; acc[0][1] = vzero; acc[1][0] = vzero; acc[1][1] = vzero;

    float4 ra[4], rb[2];
    // Prologue: tile 0 -> registers -> LDS buffer 0.
    #pragma unroll
    for (int i = 0; i < 4; ++i)
        ra[i] = *(const float4*)(X + (size_t)(m0 + sr + i * 32) * K + sc);
    #pragma unroll
    for (int i = 0; i < 2; ++i)
        rb[i] = *(const float4*)(W + (size_t)(n0 + sr + i * 32) * K + sc);
    #pragma unroll
    for (int i = 0; i < 4; ++i) {
        uint2 u; u.x = pack_bf2(ra[i].x, ra[i].y); u.y = pack_bf2(ra[i].z, ra[i].w);
        *(uint2*)&lds_a[0][(sr + i * 32) * BK + sc] = u;
    }
    #pragma unroll
    for (int i = 0; i < 2; ++i) {
        uint2 u; u.x = pack_bf2(rb[i].x, rb[i].y); u.y = pack_bf2(rb[i].z, rb[i].w);
        *(uint2*)&lds_b[0][(sr + i * 32) * BK + sc] = u;
    }
    __syncthreads();

    const int nk = K / BK;
    for (int t = 0; t < nk; ++t) {
        const int cur = t & 1, nxt = cur ^ 1;
        const int kk = (t + 1) * BK;

        // Issue next tile's global loads early (latency hides under WMMA).
        if (t + 1 < nk) {
            #pragma unroll
            for (int i = 0; i < 4; ++i)
                ra[i] = *(const float4*)(X + (size_t)(m0 + sr + i * 32) * K + kk + sc);
            #pragma unroll
            for (int i = 0; i < 2; ++i)
                rb[i] = *(const float4*)(W + (size_t)(n0 + sr + i * 32) * K + kk + sc);
            if (t + 2 < nk) {   // gfx1250 global_prefetch_b8, two tiles ahead
                if (tid < 128)
                    __builtin_prefetch(X + (size_t)(m0 + tid) * K + kk + BK, 0, 3);
                else if (tid < 192)
                    __builtin_prefetch(W + (size_t)(n0 + tid - 128) * K + kk + BK, 0, 3);
            }
        }

        // Compute from current buffer.
        v16bf afrag[2], bfrag[2];
        #pragma unroll
        for (int ms = 0; ms < 2; ++ms) {
            int row = wm * 32 + ms * 16 + ln;
            afrag[ms] = *(const v16bf*)&lds_a[cur][row * BK + koff];
        }
        #pragma unroll
        for (int ns = 0; ns < 2; ++ns) {
            int col = wn * 32 + ns * 16 + ln;
            bfrag[ns] = *(const v16bf*)&lds_b[cur][col * BK + koff];
        }
        #pragma unroll
        for (int ms = 0; ms < 2; ++ms)
            #pragma unroll
            for (int ns = 0; ns < 2; ++ns)
                acc[ms][ns] = __builtin_amdgcn_wmma_f32_16x16x32_bf16(
                    false, afrag[ms], false, bfrag[ns],
                    (short)0, acc[ms][ns], false, false);

        // Convert + store next tile into the other buffer (safe: last reader
        // of buffer `nxt` finished before the previous barrier).
        if (t + 1 < nk) {
            #pragma unroll
            for (int i = 0; i < 4; ++i) {
                uint2 u; u.x = pack_bf2(ra[i].x, ra[i].y); u.y = pack_bf2(ra[i].z, ra[i].w);
                *(uint2*)&lds_a[nxt][(sr + i * 32) * BK + sc] = u;
            }
            #pragma unroll
            for (int i = 0; i < 2; ++i) {
                uint2 u; u.x = pack_bf2(rb[i].x, rb[i].y); u.y = pack_bf2(rb[i].z, rb[i].w);
                *(uint2*)&lds_b[nxt][(sr + i * 32) * BK + sc] = u;
            }
        }
        __syncthreads();
    }

    // Epilogue: +2.0 * (T @ LB^T), then MODE-specific fuse.
    #pragma unroll
    for (int ms = 0; ms < 2; ++ms) {
        #pragma unroll
        for (int ns = 0; ns < 2; ++ns) {
            const int n = n0 + wn * 32 + ns * 16 + ln;
            const float* lbr = LB + (size_t)n * 16;
            #pragma unroll
            for (int r = 0; r < 8; ++r) {
                const int m = m0 + wm * 32 + ms * 16 + lhi * 8 + r;
                float val = acc[ms][ns][r];
                const float* tr = T + (size_t)m * 16;
                float l = 0.f;
                #pragma unroll
                for (int i = 0; i < 16; ++i) l = fmaf(tr[i], lbr[i], l);
                val = fmaf(2.0f, l, val);   // LORA_SCALE = 32/16
                const size_t oi = (size_t)m * N + n;
                if (MODE == 1) val += EX[oi];
                if (MODE == 2) { float g = EX[oi]; val *= g / (1.f + __expf(-g)); }
                out[oi] = val;
            }
        }
    }
}

// ---------------------------------------------------------------------------
// RoPE (interleaved pairs), in place on [S, H, 128].
// ---------------------------------------------------------------------------
__global__ __launch_bounds__(256) void rope_kernel(
    float* __restrict__ x, const float* __restrict__ cs,
    const float* __restrict__ sn, int H)
{
    const int idx = blockIdx.x * 256 + threadIdx.x;
    const int d   = idx & 63;
    const int t   = idx >> 6;
    const int hd  = t % H;
    const int pos = t / H;
    const float c = cs[pos * 64 + d];
    const float s = sn[pos * 64 + d];
    const size_t base = ((size_t)pos * H + hd) * 128 + 2 * d;
    const float x1 = x[base], x2 = x[base + 1];
    x[base]     = x1 * c - x2 * s;
    x[base + 1] = x1 * s + x2 * c;
}

// ---------------------------------------------------------------------------
// WMMA flash attention (causal, GQA N_REP=4).
// Block = (64-row Q tile, head): 4 waves, each owns a 16-row WMMA tile.
// Per 32-key KV tile: K staged [key][dim] (B-frag for Q@K^T), V staged
// transposed [dim][key] (B-frag for P@V), P routed C->A layout through
// wave-private LDS (in-order LDS pipe: no barrier needed). Online softmax
// with running max/sum; 1/sqrt(128) folded into Q fragments.
// ---------------------------------------------------------------------------
__global__ __launch_bounds__(128) void flash_attn_kernel(
    const float* __restrict__ q, const float* __restrict__ k,
    const float* __restrict__ v, float* __restrict__ out)
{
    __shared__ __align__(32) __bf16 lds_k[32 * 128];    // [key][dim]   8 KB
    __shared__ __align__(32) __bf16 lds_v[128 * 32];    // [dim][key]   8 KB
    __shared__ __align__(32) __bf16 lds_p[4 * 16 * 32]; // per-wave P   4 KB

    const int tid  = threadIdx.x;
    const int wv   = tid >> 5;
    const int lane = tid & 31;
    const int ln   = lane & 15;
    const int lhi  = lane >> 4;
    const int q0   = blockIdx.x * 64;
    const int h    = blockIdx.y;
    const int kvh  = h >> 2;
    const float scale = 0.08838834764831845f;   // 1/sqrt(128)
    const int klo  = lhi ? 8 : 0;
    const int koff = lhi ? 16 : 0;

    const float* qrow = q + (size_t)(q0 + wv * 16 + ln) * 4096 + h * 128;
    v16bf qf[4];
    #pragma unroll
    for (int c = 0; c < 4; ++c) {
        const float* p0 = qrow + c * 32 + koff;
        const float* p1 = qrow + c * 32 + koff + 8;
        #pragma unroll
        for (int e = 0; e < 8; ++e) {
            qf[c][e]     = bfc(p0[e] * scale);
            qf[c][e + 8] = bfc(p1[e] * scale);
        }
    }

    v8f o[8];
    const v8f vz = {0.f, 0.f, 0.f, 0.f, 0.f, 0.f, 0.f, 0.f};
    #pragma unroll
    for (int nd = 0; nd < 8; ++nd) o[nd] = vz;
    float mrun[8], lsum[8];
    #pragma unroll
    for (int r = 0; r < 8; ++r) { mrun[r] = -1e30f; lsum[r] = 0.f; }

    const int myrow_base = q0 + wv * 16 + lhi * 8;
    const float* kg = k + (size_t)kvh * 128;
    const float* vg = v + (size_t)kvh * 128;
    const int ntiles = (q0 >> 5) + 2;   // keys 0 .. q0+63

    for (int t = 0; t < ntiles; ++t) {
        const int kv0 = t * 32;
        __syncthreads();
        #pragma unroll
        for (int i = 0; i < 8; ++i) {
            int idx = tid + i * 128;
            int r = idx >> 5, c = idx & 31;
            float4 f = *(const float4*)(kg + (size_t)(kv0 + r) * 1024 + c * 4);
            uint2 u; u.x = pack_bf2(f.x, f.y); u.y = pack_bf2(f.z, f.w);
            *(uint2*)&lds_k[r * 128 + c * 4] = u;
        }
        #pragma unroll
        for (int i = 0; i < 8; ++i) {
            int idx = tid + i * 128;
            int r = idx >> 5, c = idx & 31;
            float4 f = *(const float4*)(vg + (size_t)(kv0 + r) * 1024 + c * 4);
            lds_v[(c * 4 + 0) * 32 + r] = bfc(f.x);
            lds_v[(c * 4 + 1) * 32 + r] = bfc(f.y);
            lds_v[(c * 4 + 2) * 32 + r] = bfc(f.z);
            lds_v[(c * 4 + 3) * 32 + r] = bfc(f.w);
        }
        __syncthreads();

        v8f sacc[2];
        sacc[0] = vz; sacc[1] = vz;
        #pragma unroll
        for (int ns = 0; ns < 2; ++ns) {
            #pragma unroll
            for (int c = 0; c < 4; ++c) {
                v16bf kf = *(const v16bf*)&lds_k[(ns * 16 + ln) * 128 + c * 32 + koff];
                sacc[ns] = __builtin_amdgcn_wmma_f32_16x16x32_bf16(
                    false, qf[c], false, kf, (short)0, sacc[ns], false, false);
            }
        }

        const int j0 = kv0 + ln, j1 = kv0 + 16 + ln;
        float alpha[8];
        #pragma unroll
        for (int r = 0; r < 8; ++r) {
            const int row = myrow_base + r;
            float s0 = (j0 <= row) ? sacc[0][r] : -1e30f;
            float s1 = (j1 <= row) ? sacc[1][r] : -1e30f;
            float tm = fmaxf(s0, s1);
            #pragma unroll
            for (int mvs = 1; mvs < 16; mvs <<= 1)
                tm = fmaxf(tm, __shfl_xor(tm, mvs, 32));
            float nm = fmaxf(mrun[r], tm);
            alpha[r] = __expf(mrun[r] - nm);
            mrun[r] = nm;
            float p0 = __expf(s0 - nm);
            float p1 = __expf(s1 - nm);
            float ps = p0 + p1;
            #pragma unroll
            for (int mvs = 1; mvs < 16; mvs <<= 1)
                ps += __shfl_xor(ps, mvs, 32);
            lsum[r] = lsum[r] * alpha[r] + ps;
            const int prow = (r + 8 * lhi) * 32;
            lds_p[wv * 512 + prow + ln]      = bfc(p0);
            lds_p[wv * 512 + prow + 16 + ln] = bfc(p1);
        }
        #pragma unroll
        for (int nd = 0; nd < 8; ++nd)
            #pragma unroll
            for (int r = 0; r < 8; ++r)
                o[nd][r] *= alpha[r];

        v16bf pf = *(const v16bf*)&lds_p[wv * 512 + ln * 32 + koff];

        #pragma unroll
        for (int nd = 0; nd < 8; ++nd) {
            v16bf vf = *(const v16bf*)&lds_v[(nd * 16 + ln) * 32 + koff];
            o[nd] = __builtin_amdgcn_wmma_f32_16x16x32_bf16(
                false, pf, false, vf, (short)0, o[nd], false, false);
        }
    }

    float inv[8];
    #pragma unroll
    for (int r = 0; r < 8; ++r) inv[r] = 1.f / lsum[r];
    #pragma unroll
    for (int nd = 0; nd < 8; ++nd) {
        #pragma unroll
        for (int r = 0; r < 8; ++r) {
            const int row = myrow_base + r;
            out[(size_t)row * 4096 + h * 128 + nd * 16 + ln] = o[nd][r] * inv[r];
        }
    }
}

// ---------------------------------------------------------------------------
static inline void launch_gemm(int mode, const float* X, const float* W,
                               const float* T, const float* LB, const float* EX,
                               float* out, int M, int N, int K, hipStream_t s)
{
    dim3 grid(N / BN, M / BM);
    if (mode == 0)
        gemm_lora_kernel<0><<<grid, 256, 0, s>>>(X, W, T, LB, EX, out, M, N, K);
    else if (mode == 1)
        gemm_lora_kernel<1><<<grid, 256, 0, s>>>(X, W, T, LB, EX, out, M, N, K);
    else
        gemm_lora_kernel<2><<<grid, 256, 0, s>>>(X, W, T, LB, EX, out, M, N, K);
}

extern "C" void kernel_launch(void* const* d_in, const int* in_sizes, int n_in,
                              void* d_out, int out_size, void* d_ws, size_t ws_size,
                              hipStream_t stream)
{
    (void)in_sizes; (void)n_in; (void)out_size; (void)ws_size;
    const int S = 1024, D = 4096, FF = 14336, KVD = 1024;

    const float* data   = (const float*)d_in[0];
    // d_in[1] = mask (causal; applied analytically in flash_attn_kernel)
    const float* cosb   = (const float*)d_in[2];
    const float* sinb   = (const float*)d_in[3];
    const float* anw    = (const float*)d_in[4];
    const float* wq     = (const float*)d_in[5];
    const float* wk     = (const float*)d_in[6];
    const float* wv     = (const float*)d_in[7];
    const float* wo     = (const float*)d_in[8];
    const float* wq_a   = (const float*)d_in[9];
    const float* wq_b   = (const float*)d_in[10];
    const float* wk_a   = (const float*)d_in[11];
    const float* wk_b   = (const float*)d_in[12];
    const float* wv_a   = (const float*)d_in[13];
    const float* wv_b   = (const float*)d_in[14];
    const float* wo_a   = (const float*)d_in[15];
    const float* wo_b   = (const float*)d_in[16];
    const float* fnw    = (const float*)d_in[17];
    const float* w1     = (const float*)d_in[18];
    const float* w2     = (const float*)d_in[19];
    const float* w3     = (const float*)d_in[20];
    const float* w1_a   = (const float*)d_in[21];
    const float* w1_b   = (const float*)d_in[22];
    const float* w2_a   = (const float*)d_in[23];
    const float* w2_b   = (const float*)d_in[24];
    const float* w3_a   = (const float*)d_in[25];
    const float* w3_b   = (const float*)d_in[26];

    // Workspace carve-up (~203 MB of fp32 scratch).
    float* ws    = (float*)d_ws;
    float* h     = ws;               ws += (size_t)S * D;
    float* qb    = ws;               ws += (size_t)S * D;
    float* kb    = ws;               ws += (size_t)S * KVD;
    float* vb    = ws;               ws += (size_t)S * KVD;
    float* attnb = ws;               ws += (size_t)S * D;
    float* data2 = ws;               ws += (size_t)S * D;
    float* fbuf  = ws;               ws += (size_t)S * D;
    float* gate  = ws;               ws += (size_t)S * FF;
    float* hg    = ws;               ws += (size_t)S * FF;
    float* tbuf  = ws;               ws += (size_t)S * 16;

    // ---- Attention branch ----
    rmsnorm_kernel<<<S, 256, 0, stream>>>(data, anw, h, D);

    lora_a_kernel<<<S / 16, 256, 0, stream>>>(h, wq_a, tbuf, D);
    launch_gemm(0, h, wq, tbuf, wq_b, nullptr, qb, S, D, D, stream);
    lora_a_kernel<<<S / 16, 256, 0, stream>>>(h, wk_a, tbuf, D);
    launch_gemm(0, h, wk, tbuf, wk_b, nullptr, kb, S, KVD, D, stream);
    lora_a_kernel<<<S / 16, 256, 0, stream>>>(h, wv_a, tbuf, D);
    launch_gemm(0, h, wv, tbuf, wv_b, nullptr, vb, S, KVD, D, stream);

    rope_kernel<<<(S * 32 * 64) / 256, 256, 0, stream>>>(qb, cosb, sinb, 32);
    rope_kernel<<<(S * 8 * 64) / 256, 256, 0, stream>>>(kb, cosb, sinb, 8);

    flash_attn_kernel<<<dim3(S / 64, 32), 128, 0, stream>>>(qb, kb, vb, attnb);

    lora_a_kernel<<<S / 16, 256, 0, stream>>>(attnb, wo_a, tbuf, D);
    launch_gemm(1, attnb, wo, tbuf, wo_b, data, data2, S, D, D, stream);

    // ---- FFN branch ----
    rmsnorm_kernel<<<S, 256, 0, stream>>>(data2, fnw, fbuf, D);

    lora_a_kernel<<<S / 16, 256, 0, stream>>>(fbuf, w1_a, tbuf, D);
    launch_gemm(0, fbuf, w1, tbuf, w1_b, nullptr, gate, S, FF, D, stream);
    lora_a_kernel<<<S / 16, 256, 0, stream>>>(fbuf, w3_a, tbuf, D);
    launch_gemm(2, fbuf, w3, tbuf, w3_b, gate, hg, S, FF, D, stream);   // hg = silu(gate)*up
    lora_a_kernel<<<S / 16, 256, 0, stream>>>(hg, w2_a, tbuf, FF);
    launch_gemm(1, hg, w2, tbuf, w2_b, data2, (float*)d_out, S, D, FF, stream);
}
